// LiteFlowNetCorr_70248485093645
// MI455X (gfx1250) — matrix operand order repaired and checked by reading, not compiled, not run on hardware
//
#include <hip/hip_runtime.h>
#include <hip/hip_bf16.h>

// ---------------------------------------------------------------------------
// LiteFlowNet correlation + conv head for MI455X (gfx1250, wave32, WMMA).
// All heavy math runs through v_wmma_f32_16x16x32_f16 (f16 in, f32 accum).
// - Activations in spatially zero-padded NHWC f16 -> all hot loads unconditional.
// - Conv weights staged per-block in LDS (8 waves share one N-tile) -> 8x less
//   global weight traffic; B-fragments come from ds_load.
// - Conv K-loop software-pipelined one chunk ahead; corr preloads all B frags.
// ---------------------------------------------------------------------------

typedef __attribute__((ext_vector_type(16))) _Float16 v16h;
typedef __attribute__((ext_vector_type(8)))  _Float16 v8h;
typedef __attribute__((ext_vector_type(8)))  float    v8f;
typedef __attribute__((ext_vector_type(4)))  unsigned int v4u;

static __device__ __forceinline__ v16h cat8(v8h lo, v8h hi) {
    return __builtin_shufflevector(lo, hi, 0,1,2,3,4,5,6,7,8,9,10,11,12,13,14,15);
}
// B-fragment: 16 contiguous f16 (lane = N column, K = kbase..kbase+15)
static __device__ __forceinline__ v16h load16(const _Float16* p) {
    return cat8(*(const v8h*)p, *(const v8h*)(p + 8));
}
// A-fragment: elements 0..15 = p[0..15]
// (contiguous K layout: lane L holds row M=L&15, K in {kb16..kb16+15})
static __device__ __forceinline__ v16h loadA_gap(const _Float16* p) {
    return cat8(*(const v8h*)p, *(const v8h*)(p + 8));
}

#define WMMA_F16(A, B, C) \
    __builtin_amdgcn_wmma_f32_16x16x32_f16(false, (A), false, (B), (short)0, (C), false, false)

// Padded geometry
#define HD 128
#define WD 224
#define NWT (WD / 16)    // 14 W-tiles
#define HP (HD + 2)      // 130  (1-px halo for 3x3 convs)
#define WP (WD + 2)      // 226
#define HP2 (HD + 8)     // 136  (4-row halo for correlation dy)
#define WP2 (WD + 16)    // 240  (8-col halo for correlation dx windows)

// ---------------------------------------------------------------------------
// 16B-granular zero fill (halo + pad channels become zero once per call).
// ---------------------------------------------------------------------------
__global__ void __launch_bounds__(256)
fill_zero16(v4u* __restrict__ dst, int total16) {
    int i = blockIdx.x * blockDim.x + threadIdx.x;
    if (i >= total16) return;
    v4u z = {};
    dst[i] = z;
}

// ---------------------------------------------------------------------------
// fp32 NCHW -> f16 NHWC into padded interior.
// ---------------------------------------------------------------------------
__global__ void __launch_bounds__(256)
pack_nchw_to_nhwc_pad(const float* __restrict__ src, _Float16* __restrict__ dst,
                      int C, int dstC, int Hp, int Wp, int padH, int padW, int total) {
    int i = blockIdx.x * blockDim.x + threadIdx.x;
    if (i >= total) return;
    int w = i % WD;
    int h = (i / WD) % HD;
    int c = (i / (WD * HD)) % C;
    int b = i / (WD * HD * C);
    size_t o = ((size_t)(b * Hp + h + padH) * Wp + (w + padW)) * dstC + c;
    dst[o] = (_Float16)src[i];
}

// ---------------------------------------------------------------------------
// Pack OIHW fp32 weights into WMMA B-fragment order, ct-major so the slab for
// one N-tile is contiguous (block = ct*(9*KC) + t*KC + c32, 512 f16/block):
//   (lane,e): cout = ct*16 + (lane&15), cin = c32*32 + (lane>>4)*16 + e
// Zero-pads cin (209->224) and cout (2->16).
// ---------------------------------------------------------------------------
__global__ void __launch_bounds__(256)
pack_weights_kernel(const float* __restrict__ w, _Float16* __restrict__ dst,
                    int cinReal, int cinPad, int coutReal, int total) {
    int i = blockIdx.x * blockDim.x + threadIdx.x;
    if (i >= total) return;
    int e    = i & 15;
    int lane = (i >> 4) & 31;
    int blk  = i >> 9;
    int kc   = cinPad >> 5;
    int c32  = blk % kc;
    int t    = (blk / kc) % 9;
    int ct   = blk / (kc * 9);
    int cout = ct * 16 + (lane & 15);
    int cin  = c32 * 32 + (lane >> 4) * 16 + e;
    float v = 0.0f;
    if (cout < coutReal && cin < cinReal) {
        int ky = t / 3, kx = t % 3;
        v = w[(((size_t)cout * cinReal + cin) * 3 + ky) * 3 + kx];
    }
    dst[i] = (_Float16)v;
}

// ---------------------------------------------------------------------------
// Correlation via WMMA: per (b,h,16-pixel tile), for each dy in [-4,4]:
//   out[w1,w2] = sum_c x1[c,w1]*x2[c,w2]  (16x128 GEMM against two 16-col
//   windows of x2); extract diagonal band w2-w1 in [-4,4] -> 81 channels.
// x2p is spatially padded: all loads unconditional; dy advance is a constant
// pointer increment; all 8 B-fragments preloaded before the 8 WMMAs.
// ---------------------------------------------------------------------------
__global__ void __launch_bounds__(256)
corr_wmma_kernel(_Float16* __restrict__ h0, const _Float16* __restrict__ x2p, int nTiles) {
    int wave = (int)((blockIdx.x * blockDim.x + threadIdx.x) >> 5);
    if (wave >= nTiles) return;             // wave-uniform: EXEC all-ones below
    int wt = wave % NWT;
    int h  = (wave / NWT) % HD;
    int b  = wave / (NWT * HD);
    int w0 = wt * 16;

    int lane  = threadIdx.x & 31;
    int rowM  = lane & 15;
    int half  = lane >> 4;
    int base8 = half * 8;
    int kb16  = half * 16;
    int col   = lane & 15;

    // A = x1 tile (first 128 channels of h0 interior), hoisted across all dy.
    const _Float16* pA =
        h0 + ((size_t)(b * HP + h + 1) * WP + (w0 + rowM + 1)) * 224;
    v16h A[4];
#pragma unroll
    for (int c = 0; c < 4; ++c)
        A[c] = loadA_gap(pA + c * 32 + kb16);

    _Float16* pO = h0 + ((size_t)(b * HP + h + 1) * WP + (w0 + 1)) * 224;

    // padded base row: (h + dy - 4) + 4 = h + dy ; padded col: w2 + 8
    const _Float16* pB0 =
        x2p + ((size_t)(b * HP2 + h) * WP2 + (w0 + col)) * 128 + kb16;       // w2 = w0-8+col
    const _Float16* pB1 = pB0 + (size_t)16 * 128;                            // w2 = w0+8+col

    for (int dy = 0; dy < 9; ++dy) {
        v16h B0[4], B1[4];
#pragma unroll
        for (int c = 0; c < 4; ++c) {
            B0[c] = load16(pB0 + c * 32);
            B1[c] = load16(pB1 + c * 32);
        }
        v8f acc0 = {}; v8f acc1 = {};
#pragma unroll
        for (int c = 0; c < 4; ++c) {
            acc0 = WMMA_F16(A[c], B0[c], acc0);
            acc1 = WMMA_F16(A[c], B1[c], acc1);
        }
#pragma unroll
        for (int r = 0; r < 8; ++r) {
            int w1 = r + base8;
            int dx0 = col - 8 - w1;       // window [w0-8, w0+8)
            if (dx0 >= -4 && dx0 <= 4)
                pO[(size_t)w1 * 224 + 128 + dy * 9 + (dx0 + 4)] =
                    (_Float16)(acc0[r] * 0.0078125f);   // mean over C=128
            int dx1 = col + 8 - w1;       // window [w0+8, w0+24)
            if (dx1 >= -4 && dx1 <= 4)
                pO[(size_t)w1 * 224 + 128 + dy * 9 + (dx1 + 4)] =
                    (_Float16)(acc1[r] * 0.0078125f);
        }
        pB0 += (size_t)WP2 * 128;         // next dy row
        pB1 += (size_t)WP2 * 128;
    }
}

// ---------------------------------------------------------------------------
// 3x3 SAME conv as implicit GEMM: M=16 W-pixels, N=16 out-channels,
// K = CIN*9 in 32-channel chunks. 8 waves/block share one N-tile: its weight
// slab (9*KC*512 f16) is staged in LDS once, B-fragments come from ds_load.
// A-loads are software-pipelined one K-chunk ahead. Exact grid (1792 % 8 == 0),
// so no tail guard and the block barrier is safe.
// ---------------------------------------------------------------------------
template <int CIN, bool LEAKY, bool FINAL>
__global__ void __launch_bounds__(256)
conv3x3_wmma_kernel(const _Float16* __restrict__ in, const _Float16* __restrict__ wpk,
                    const float* __restrict__ bias, _Float16* __restrict__ out,
                    float* __restrict__ out32, int coutPad, int coutReal, int nCT) {
    constexpr int KC   = CIN / 32;
    constexpr int NK   = 9 * KC;          // K-chunks total
    constexpr int SLAB = NK * 512;        // f16 elements per N-tile slab
    __shared__ _Float16 smem[SLAB];

    const int GRP = (HD * NWT) / 8;       // 224 groups of 8 wave-tiles
    int g  = blockIdx.x % GRP;
    int ct = (blockIdx.x / GRP) % nCT;
    int b  = blockIdx.x / (GRP * nCT);

    // Cooperative weight slab load: global -> LDS (16B per thread-iteration).
    {
        const v4u* gs = (const v4u*)(wpk + (size_t)ct * SLAB);
        v4u* ls = (v4u*)smem;
#pragma unroll 4
        for (int i = threadIdx.x; i < SLAB / 8; i += 256) ls[i] = gs[i];
    }
    __syncthreads();

    int f  = g * 8 + (int)(threadIdx.x >> 5);
    int wt = f % NWT;
    int h  = f / NWT;
    int w0 = wt * 16;

    int lane  = threadIdx.x & 31;
    int rowM  = lane & 15;
    int half  = lane >> 4;
    int base8 = half * 8;
    int colN  = lane & 15;

    // A address for flattened K-chunk kk (t = kk/KC taps, c = kk%KC chunks).
    auto addrA = [&](int kk) -> const _Float16* {
        int t = kk / KC, c = kk % KC;
        int ty = t / 3, tx = t % 3;       // padded offsets 0..2 (interior at +1)
        return in + ((size_t)(b * HP + h + ty) * WP + (w0 + rowM + tx)) * CIN
                  + half * 16 + c * 32;
    };

    v8f acc = {};
    v16h aCur = loadA_gap(addrA(0));
#pragma unroll
    for (int kk = 0; kk < NK; ++kk) {
        v16h aNext = aCur;
        if (kk + 1 < NK) aNext = loadA_gap(addrA(kk + 1));   // prefetch next chunk
        v16h bf = load16(&smem[kk * 512 + lane * 16]);        // ds_load from slab
        acc = WMMA_F16(aCur, bf, acc);
        aCur = aNext;
    }

    int co = ct * 16 + colN;
    float bv = (co < coutReal) ? bias[co] : 0.0f;
#pragma unroll
    for (int r = 0; r < 8; ++r) {
        float v = acc[r] + bv;
        if (LEAKY) v = (v >= 0.0f) ? v : 0.1f * v;
        int m = r + base8;
        if (FINAL) {
            if (co < 2)
                out32[((size_t)(b * 2 + co) * HD + h) * WD + w0 + m] = v;
        } else {
            out[((size_t)(b * HP + h + 1) * WP + (w0 + m + 1)) * coutPad + co] =
                (_Float16)v;
        }
    }
}

// ---------------------------------------------------------------------------
extern "C" void kernel_launch(void* const* d_in, const int* in_sizes, int n_in,
                              void* d_out, int out_size, void* d_ws, size_t ws_size,
                              hipStream_t stream) {
    (void)in_sizes; (void)n_in; (void)out_size; (void)ws_size;
    const int Bn = 8, Cc = 128;
    const size_t PIXP  = (size_t)Bn * HP * WP;    // padded conv pixels
    const size_t PIXP2 = (size_t)Bn * HP2 * WP2;  // padded corr pixels

    const float* x1 = (const float*)d_in[0];
    const float* x2 = (const float*)d_in[1];
    const float* w1 = (const float*)d_in[2]; const float* b1 = (const float*)d_in[3];
    const float* w2 = (const float*)d_in[4]; const float* b2 = (const float*)d_in[5];
    const float* w3 = (const float*)d_in[6]; const float* b3 = (const float*)d_in[7];
    const float* w4 = (const float*)d_in[8]; const float* b4 = (const float*)d_in[9];
    float* out = (float*)d_out;

    // Workspace carve-up (~248 MB), 256B aligned regions.
    char* ws = (char*)d_ws;
    size_t off = 0;
    auto carve = [&](size_t bytes) -> void* {
        void* p = ws + off;
        off = (off + bytes + 255) & ~(size_t)255;
        return p;
    };
    const size_t h0_bytes = PIXP  * 224 * 2;  // x1 | corr | zero-pad channels
    const size_t x2_bytes = PIXP2 * 128 * 2;
    const size_t h1_bytes = PIXP  * 64 * 2;
    const size_t h3_bytes = PIXP  * 32 * 2;
    _Float16* h0p = (_Float16*)carve(h0_bytes);
    _Float16* x2p = (_Float16*)carve(x2_bytes);
    _Float16* h1p = (_Float16*)carve(h1_bytes);
    _Float16* h2p = (_Float16*)carve(h1_bytes);
    _Float16* h3p = (_Float16*)carve(h3_bytes);
    _Float16* wp1 = (_Float16*)carve((size_t)4 * 9 * 7 * 512 * 2);
    _Float16* wp2 = (_Float16*)carve((size_t)4 * 9 * 2 * 512 * 2);
    _Float16* wp3 = (_Float16*)carve((size_t)2 * 9 * 2 * 512 * 2);
    _Float16* wp4 = (_Float16*)carve((size_t)1 * 9 * 1 * 512 * 2);

    // 1) Zero fill padded buffers (halo + pad channels).
    auto fill = [&](void* p, size_t bytes) {
        int t16 = (int)(bytes / 16);
        fill_zero16<<<(t16 + 255) / 256, 256, 0, stream>>>((v4u*)p, t16);
    };
    fill(h0p, h0_bytes);
    fill(x2p, x2_bytes);
    fill(h1p, h1_bytes);
    fill(h2p, h1_bytes);
    fill(h3p, h3_bytes);

    // 2) Layout conversion fp32 NCHW -> f16 NHWC (padded interiors).
    {
        int total = Bn * Cc * HD * WD;
        int blocks = (total + 255) / 256;
        pack_nchw_to_nhwc_pad<<<blocks, 256, 0, stream>>>(x1, h0p, Cc, 224, HP, WP, 1, 1, total);
        pack_nchw_to_nhwc_pad<<<blocks, 256, 0, stream>>>(x2, x2p, Cc, 128, HP2, WP2, 4, 8, total);
    }

    // 3) Weight packing into ct-major B-fragment slabs (K-padded, N-padded).
    pack_weights_kernel<<<(129024 + 255) / 256, 256, 0, stream>>>(w1, wp1, 209, 224, 64, 129024);
    pack_weights_kernel<<<(36864  + 255) / 256, 256, 0, stream>>>(w2, wp2,  64,  64, 64, 36864);
    pack_weights_kernel<<<(18432  + 255) / 256, 256, 0, stream>>>(w3, wp3,  64,  64, 32, 18432);
    pack_weights_kernel<<<(4608   + 255) / 256, 256, 0, stream>>>(w4, wp4,  32,  32,  2, 4608);

    // 4) Correlation (writes h0p channels 128..208).
    {
        int nT = Bn * HD * NWT;
        corr_wmma_kernel<<<(nT * 32 + 255) / 256, 256, 0, stream>>>(h0p, x2p, nT);
    }

    // 5) Conv head. Grid = Bn * nCT * 224 blocks of 8 waves (exact, no tail).
    {
        const int GRP = (HD * NWT) / 8;   // 224
        conv3x3_wmma_kernel<224, true, false><<<Bn * 4 * GRP, 256, 0, stream>>>(
            h0p, wp1, b1, h1p, nullptr, 64, 64, 4);
        conv3x3_wmma_kernel<64, true, false><<<Bn * 4 * GRP, 256, 0, stream>>>(
            h1p, wp2, b2, h2p, nullptr, 64, 64, 4);
        conv3x3_wmma_kernel<64, true, false><<<Bn * 2 * GRP, 256, 0, stream>>>(
            h2p, wp3, b3, h3p, nullptr, 32, 32, 2);
        conv3x3_wmma_kernel<32, false, true><<<Bn * 1 * GRP, 256, 0, stream>>>(
            h3p, wp4, b4, nullptr, out, 16, 2, 1);
    }
}
